// HawkesKT_83330955477181
// MI455X (gfx1250) — matrix operand order repaired and checked by reading, not compiled, not run on hardware
//
#include <hip/hip_runtime.h>
#include <hip/hip_bf16.h>
#include <math.h>

// HawkesKT on MI455X (gfx1250, wave32, WMMA f16->f32)
//
// shapes: B=64, L=1024, D=64, N_SKILLS=1000
// out[b, j] = sigmoid( problem_base[problems[b,j]] + skill_base[skills[b,j]]
//                      + sum_{i<j} alphas[b,i,j] * exp(-clip(betas[b,i,j]+1,0,10) * dlog(i,j)) )
// alphas[b,i,j] = dot(alpha_inter[inters[b,i]], alpha_skill[skills[b,j]])
// betas  likewise with beta_* tables.
// dlog(i,j) = log(|t_i - t_j| + 1e-10)/log(5),  t = times/1000.

typedef __attribute__((ext_vector_type(16))) _Float16 v16h;
typedef __attribute__((ext_vector_type(8)))  float    v8f;

#define HKT_B 64
#define HKT_L 1024
#define HKT_D 64
#define HKT_NSKILLS 1000

__global__ __launch_bounds__(32)
void HawkesKT_83330955477181_kernel(
    const int*   __restrict__ skills,       // (B,L)
    const int*   __restrict__ problems,     // (B,L)
    const int*   __restrict__ times,        // (B,L)
    const int*   __restrict__ labels,       // (B,L)
    const float* __restrict__ problem_base, // (N_PROBLEMS,1)
    const float* __restrict__ skill_base,   // (N_SKILLS,1)
    const float* __restrict__ alpha_inter,  // (2*N_SKILLS, D)
    const float* __restrict__ alpha_skill,  // (N_SKILLS, D)
    const float* __restrict__ beta_inter,   // (2*N_SKILLS, D)
    const float* __restrict__ beta_skill,   // (N_SKILLS, D)
    float*       __restrict__ out)          // (B,L)
{
    const int lane = threadIdx.x;        // 0..31
    const int n    = lane & 15;          // column within j-tile (B/C/D layout)
    const int g    = lane >> 4;          // lane-group (0 or 1)
    const int jt   = blockIdx.x;         // j tile index, 0..L/16-1
    const int b    = blockIdx.y;         // batch
    const int j0   = jt * 16;
    const int j    = j0 + n;

    const int rowbase = b * HKT_L;

    // ---------------- Build B fragments for this j-tile (shared over all i-tiles) --------
    // 16-bit B 32x16 layout (wave32): slots 0..7 -> K = g*8 + p,
    // slots 8..15 -> K = 16 + g*8 + (p-8). Two K=32 steps cover D=64.
    const int   sj      = skills[rowbase + j];
    const float* at_row = alpha_skill + sj * HKT_D;
    const float* bt_row = beta_skill  + sj * HKT_D;

    v16h Ba0, Ba1, Bb0, Bb1;   // alpha/beta targets, k-step 0 (d 0..31) and 1 (d 32..63)
#pragma unroll
    for (int p = 0; p < 16; ++p) {
        const int d0 = (p < 8) ? (g * 8 + p) : (16 + g * 8 + (p - 8));        // k-step 0
        const int d1 = 32 + d0;   // k-step 1
        Ba0[p] = (_Float16)at_row[d0];
        Ba1[p] = (_Float16)at_row[d1];
        Bb0[p] = (_Float16)bt_row[d0];
        Bb1[p] = (_Float16)bt_row[d1];
    }

    // target time for this lane's column
    const float tj = (float)times[rowbase + j] * 1.0e-3f;

    const float inv_ln5 = 0.62133493f;    // 1/ln(5)

    float colsum = 0.0f;                  // partial sum over this lane's m's for column n

    // ---------------- Loop over causal i-tiles ----------------
    for (int it = 0; it <= jt; ++it) {
        const int i0 = it * 16;
        const int m  = n;                 // A layout: lanes 0-15 and 16-31 both map m = lane%16
        const int i_row = i0 + m;

        const int si    = skills[rowbase + i_row];
        const int li    = labels[rowbase + i_row];
        const int inter = si + li * HKT_NSKILLS;

        const float* ai_row = alpha_inter + inter * HKT_D;
        const float* bi_row = beta_inter  + inter * HKT_D;

        // 16-bit A 16x32 layout (wave32):
        //   slots 0..7  -> K = g*8 + p
        //   slots 8..15 -> K = 16 + g*8 + p
        v16h Aa0, Aa1, Ab0, Ab1;
#pragma unroll
        for (int p = 0; p < 8; ++p) {
            const int dA = g * 8 + p;
            Aa0[p]     = (_Float16)ai_row[dA];          // k-step 0, K in [0,16)
            Aa0[8 + p] = (_Float16)ai_row[16 + dA];     // k-step 0, K in [16,32)
            Aa1[p]     = (_Float16)ai_row[32 + dA];     // k-step 1
            Aa1[8 + p] = (_Float16)ai_row[48 + dA];
            Ab0[p]     = (_Float16)bi_row[dA];
            Ab0[8 + p] = (_Float16)bi_row[16 + dA];
            Ab1[p]     = (_Float16)bi_row[32 + dA];
            Ab1[8 + p] = (_Float16)bi_row[48 + dA];
        }

        // ---- WMMA: 16x16 tiles of alphas and betas, f32 accumulate over K=64 ----
        v8f ca = {};
        ca = __builtin_amdgcn_wmma_f32_16x16x32_f16(false, Aa0, false, Ba0,
                                                    (short)0, ca, false, false);
        ca = __builtin_amdgcn_wmma_f32_16x16x32_f16(false, Aa1, false, Ba1,
                                                    (short)0, ca, false, false);
        v8f cb = {};
        cb = __builtin_amdgcn_wmma_f32_16x16x32_f16(false, Ab0, false, Bb0,
                                                    (short)0, cb, false, false);
        cb = __builtin_amdgcn_wmma_f32_16x16x32_f16(false, Ab1, false, Bb1,
                                                    (short)0, cb, false, false);

        // ---- Elementwise epilogue on the 16x16 tile ----
        // C/D layout: slot v -> m = v + 8*g, column n = lane%16.
#pragma unroll
        for (int v = 0; v < 8; ++v) {
            const int mi = v + 8 * g;
            const int i  = i0 + mi;

            const float ti  = (float)times[rowbase + i] * 1.0e-3f;
            float dt        = fabsf(ti - tj);
            float dlog      = __logf(dt + 1.0e-10f) * inv_ln5;

            float beta = cb[v] + 1.0f;
            beta = fminf(fmaxf(beta, 0.0f), 10.0f);

            float cross = ca[v] * __expf(-beta * dlog);
            colsum += (i < j) ? cross : 0.0f;
        }
    }

    // lanes l and l+16 cover m=0..7 / m=8..15 of the same column n -> pairwise reduce
    colsum += __shfl_xor(colsum, 16, 32);

    if (g == 0) {
        const float h = problem_base[problems[rowbase + j]]
                      + skill_base[skills[rowbase + j]]
                      + colsum;
        out[rowbase + j] = 1.0f / (1.0f + __expf(-h));
    }
}

extern "C" void kernel_launch(void* const* d_in, const int* in_sizes, int n_in,
                              void* d_out, int out_size, void* d_ws, size_t ws_size,
                              hipStream_t stream) {
    const int*   skills       = (const int*)  d_in[0];
    const int*   problems     = (const int*)  d_in[1];
    const int*   times        = (const int*)  d_in[2];
    const int*   labels       = (const int*)  d_in[3];
    const float* problem_base = (const float*)d_in[4];
    const float* skill_base   = (const float*)d_in[5];
    const float* alpha_inter  = (const float*)d_in[6];
    const float* alpha_skill  = (const float*)d_in[7];
    const float* beta_inter   = (const float*)d_in[8];
    const float* beta_skill   = (const float*)d_in[9];
    float* out = (float*)d_out;

    dim3 grid(HKT_L / 16, HKT_B);   // 64 j-tiles x 64 batches = 4096 waves
    dim3 block(32);                 // one wave32 per workgroup
    HawkesKT_83330955477181_kernel<<<grid, block, 0, stream>>>(
        skills, problems, times, labels, problem_base, skill_base,
        alpha_inter, alpha_skill, beta_inter, beta_skill, out);
}
